// MultiHeadAttention_35467839930683
// MI455X (gfx1250) — matrix operand order repaired and checked by reading, not compiled, not run on hardware
//
#include <hip/hip_runtime.h>

// ---------------- problem constants ----------------
#define BB      2
#define LL      2048
#define DMODEL  2048
#define E3      6144          // 3*DMODEL
#define NH      16
#define DH      128
#define NTOK    (BB*LL)       // 4096

typedef __attribute__((ext_vector_type(16))) __bf16 bf16x16;
typedef __attribute__((ext_vector_type(8)))  __bf16 bf16x8;
typedef __attribute__((ext_vector_type(8)))  float  f32x8;

union BF16Frag { bf16x16 v; bf16x8 h[2]; };

__device__ __forceinline__ f32x8 wmma_bf16(bf16x16 a, bf16x16 b, f32x8 c) {
    return __builtin_amdgcn_wmma_f32_16x16x32_bf16(
        /*neg_a=*/false, a, /*neg_b=*/false, b,
        /*c_mod=*/(short)0, c, /*reuse_a=*/false, /*reuse_b=*/false);
}

// A-fragment of a 16x32 bf16 tile, row-major source with element stride `stride`.
// lane l holds row (l&15); K runs [(l>>4)*8, +8) and [+16, +24) -> two b128 loads.
__device__ __forceinline__ bf16x16 load_a_frag(const __bf16* base, int stride) {
    const int lane = threadIdx.x & 31;
    const int row  = lane & 15;
    const int kb   = (lane >> 4) * 8;
    BF16Frag f;
    const __bf16* p = base + (size_t)row * stride + kb;
    f.h[0] = *(const bf16x8*)(p);
    f.h[1] = *(const bf16x8*)(p + 16);
    return f.v;
}

// B-fragment of a 32x16 bf16 tile where source is laid out "one row per output
// column, contiguous in K" (i.e. W[e,d] row-major, or V transposed [dh][L]).
// lane l holds col (l&15); K runs [(l>>4)*8, +8) and [+16, +24) -> two b128 loads.
__device__ __forceinline__ bf16x16 load_b_frag(const __bf16* base, int stride) {
    const int lane = threadIdx.x & 31;
    const int n    = lane & 15;
    const int kb   = (lane >> 4) * 8;
    BF16Frag f;
    const __bf16* p = base + (size_t)n * stride + kb;
    f.h[0] = *(const bf16x8*)(p);
    f.h[1] = *(const bf16x8*)(p + 16);
    return f.v;
}

// ---------------- fp32 -> bf16 convert ----------------
__global__ void cvt_f32_bf16_kernel(const float* __restrict__ in,
                                    __bf16* __restrict__ out, int n) {
    int i = blockIdx.x * blockDim.x + threadIdx.x;
    if (i < n) out[i] = (__bf16)in[i];
}

// ---------------- QKV GEMM: [NTOK,DMODEL] x [E3,DMODEL]^T + b ----------------
// Each wave: 32(M) x 64(N) block of C -> 8 C-frags, 8 WMMAs per K-step of 32.
// Block = 4 waves covering 256 output columns.
// Epilogue scatters bf16 into Q[b,h,l,dh], K[b,h,l,dh], Vt[b,h,dh,l].
__device__ __forceinline__ void store_qkv_tile(const f32x8 c, int m0, int n0,
                                               const float* __restrict__ bias,
                                               __bf16* __restrict__ Q,
                                               __bf16* __restrict__ K,
                                               __bf16* __restrict__ Vt) {
    const int lane = threadIdx.x & 31;
    const int n    = n0 + (lane & 15);
    const int which = n >> 11;          // 0:q 1:k 2:v
    const int h     = (n >> 7) & (NH - 1);
    const int dh    = n & (DH - 1);
    const float bv  = bias[n];
#pragma unroll
    for (int i = 0; i < 8; ++i) {
        const int m = m0 + i + ((lane >> 4) << 3);
        const int b = m >> 11;
        const int l = m & (LL - 1);
        const __bf16 o = (__bf16)(c[i] + bv);
        if (which == 0)
            Q[(((size_t)(b*NH + h))*LL + l)*DH + dh] = o;
        else if (which == 1)
            K[(((size_t)(b*NH + h))*LL + l)*DH + dh] = o;
        else
            Vt[(((size_t)(b*NH + h))*DH + dh)*LL + l] = o;
    }
}

__global__ void qkv_gemm_kernel(const __bf16* __restrict__ X,
                                const __bf16* __restrict__ W,
                                const float*  __restrict__ bias,
                                __bf16* __restrict__ Q,
                                __bf16* __restrict__ K,
                                __bf16* __restrict__ Vt) {
    const int wave = threadIdx.x >> 5;
    const int n0 = blockIdx.x * 256 + wave * 64;
    const int m0 = blockIdx.y * 32;
    f32x8 c[2][4];
#pragma unroll
    for (int i = 0; i < 2; ++i)
#pragma unroll
        for (int j = 0; j < 4; ++j) c[i][j] = f32x8{};

    const __bf16* Arow[2] = { X + (size_t)m0 * DMODEL,
                              X + (size_t)(m0 + 16) * DMODEL };
    const __bf16* Brow[4] = { W + (size_t)(n0     ) * DMODEL,
                              W + (size_t)(n0 + 16) * DMODEL,
                              W + (size_t)(n0 + 32) * DMODEL,
                              W + (size_t)(n0 + 48) * DMODEL };
    for (int k = 0; k < DMODEL; k += 32) {
        __builtin_prefetch(Arow[0] + k + 128, 0, 3);
        __builtin_prefetch(Brow[0] + k + 128, 0, 3);
        __builtin_prefetch(Brow[2] + k + 128, 0, 3);
        bf16x16 a[2], b[4];
#pragma unroll
        for (int i = 0; i < 2; ++i) a[i] = load_a_frag(Arow[i] + k, DMODEL);
#pragma unroll
        for (int j = 0; j < 4; ++j) b[j] = load_b_frag(Brow[j] + k, DMODEL);
#pragma unroll
        for (int i = 0; i < 2; ++i)
#pragma unroll
            for (int j = 0; j < 4; ++j)
                c[i][j] = wmma_bf16(a[i], b[j], c[i][j]);
    }
#pragma unroll
    for (int i = 0; i < 2; ++i)
#pragma unroll
        for (int j = 0; j < 4; ++j)
            store_qkv_tile(c[i][j], m0 + i * 16, n0 + j * 16, bias, Q, K, Vt);
}

// ---------------- flash attention: one wave per 16-query tile ----------------
__global__ void attn_kernel(const __bf16* __restrict__ Q,
                            const __bf16* __restrict__ Kmat,
                            const __bf16* __restrict__ Vt,
                            __bf16* __restrict__ O) {
    __shared__ __bf16 Plds[4][16][32];    // per-wave P tile for C->A relayout
    const int lane = threadIdx.x & 31;
    const int wave = threadIdx.x >> 5;
    const int qt = blockIdx.x * 4 + wave;             // 0..127 query tile
    const int h  = blockIdx.y;
    const int b  = blockIdx.z;
    const size_t bh = (size_t)(b * NH + h);
    const __bf16* Qb = Q    + bh * LL * DH + (size_t)qt * 16 * DH;
    const __bf16* Kb = Kmat + bh * LL * DH;
    const __bf16* Vb = Vt   + bh * DH * LL;

    bf16x16 qf[4];
#pragma unroll
    for (int d = 0; d < 4; ++d) qf[d] = load_a_frag(Qb + d * 32, DH);

    f32x8 o[8];
#pragma unroll
    for (int j = 0; j < 8; ++j) o[j] = f32x8{};
    float rmax[8], rsum[8];
#pragma unroll
    for (int i = 0; i < 8; ++i) { rmax[i] = -1e30f; rsum[i] = 0.0f; }
    const float scl = 0.08838834764831845f;           // 1/sqrt(128)

    for (int kc = 0; kc < LL; kc += 32) {
        f32x8 s0{}, s1{};
#pragma unroll
        for (int d = 0; d < 4; ++d) {
            bf16x16 kb0 = load_b_frag(Kb + (size_t)kc        * DH + d * 32, DH);
            bf16x16 kb1 = load_b_frag(Kb + (size_t)(kc + 16) * DH + d * 32, DH);
            s0 = wmma_bf16(qf[d], kb0, s0);
            s1 = wmma_bf16(qf[d], kb1, s1);
        }
        float alpha[8];
#pragma unroll
        for (int i = 0; i < 8; ++i) {
            s0[i] *= scl; s1[i] *= scl;
            float t = fmaxf(s0[i], s1[i]);
#pragma unroll
            for (int mask = 1; mask <= 8; mask <<= 1)
                t = fmaxf(t, __shfl_xor(t, mask, 32));
            const float mnew = fmaxf(rmax[i], t);
            const float a  = __expf(rmax[i] - mnew);
            const float p0 = __expf(s0[i] - mnew);
            const float p1 = __expf(s1[i] - mnew);
            s0[i] = p0; s1[i] = p1;
            float rs = p0 + p1;
#pragma unroll
            for (int mask = 1; mask <= 8; mask <<= 1)
                rs += __shfl_xor(rs, mask, 32);
            rsum[i] = rsum[i] * a + rs;
            rmax[i] = mnew;
            alpha[i] = a;
        }
#pragma unroll
        for (int j = 0; j < 8; ++j)
#pragma unroll
            for (int i = 0; i < 8; ++i) o[j][i] *= alpha[i];

        // P (C-layout) -> LDS -> A-layout
        {
            const int r  = (lane >> 4) * 8;
            const int nn = lane & 15;
#pragma unroll
            for (int i = 0; i < 8; ++i) {
                Plds[wave][r + i][nn]      = (__bf16)s0[i];
                Plds[wave][r + i][nn + 16] = (__bf16)s1[i];
            }
        }
        __syncthreads();
        bf16x16 pa;
        {
            const int row = lane & 15;
            const int kb  = (lane >> 4) * 8;
            BF16Frag f;
            f.h[0] = *(const bf16x8*)(&Plds[wave][row][kb]);
            f.h[1] = *(const bf16x8*)(&Plds[wave][row][kb + 16]);
            pa = f.v;
        }
        __syncthreads();
#pragma unroll
        for (int j = 0; j < 8; ++j) {
            bf16x16 vb = load_b_frag(Vb + (size_t)(j * 16) * LL + kc, LL);
            o[j] = wmma_bf16(pa, vb, o[j]);
        }
    }

    // epilogue: normalize, store bf16 to attn_out[token, h*128 + dh]
    const int nn = lane & 15;
#pragma unroll
    for (int i = 0; i < 8; ++i) {
        const int m = qt * 16 + i + ((lane >> 4) << 3);
        const size_t tok = (size_t)b * LL + m;
        const float inv = 1.0f / rsum[i];
#pragma unroll
        for (int j = 0; j < 8; ++j)
            O[tok * DMODEL + h * DH + j * 16 + nn] = (__bf16)(o[j][i] * inv);
    }
}

// ---------------- output projection: fp32 result ----------------
__device__ __forceinline__ void store_out_tile(const f32x8 c, int m0, int n0,
                                               const float* __restrict__ bias,
                                               float* __restrict__ out) {
    const int lane = threadIdx.x & 31;
    const int n = n0 + (lane & 15);
    const float bv = bias[n];
#pragma unroll
    for (int i = 0; i < 8; ++i) {
        const int m = m0 + i + ((lane >> 4) << 3);
        out[(size_t)m * DMODEL + n] = c[i] + bv;
    }
}

__global__ void proj_gemm_kernel(const __bf16* __restrict__ A,
                                 const __bf16* __restrict__ W,
                                 const float*  __restrict__ bias,
                                 float* __restrict__ out) {
    const int wave = threadIdx.x >> 5;
    const int n0 = blockIdx.x * 256 + wave * 64;
    const int m0 = blockIdx.y * 32;
    f32x8 c[2][4];
#pragma unroll
    for (int i = 0; i < 2; ++i)
#pragma unroll
        for (int j = 0; j < 4; ++j) c[i][j] = f32x8{};

    const __bf16* Arow[2] = { A + (size_t)m0 * DMODEL,
                              A + (size_t)(m0 + 16) * DMODEL };
    const __bf16* Brow[4] = { W + (size_t)(n0     ) * DMODEL,
                              W + (size_t)(n0 + 16) * DMODEL,
                              W + (size_t)(n0 + 32) * DMODEL,
                              W + (size_t)(n0 + 48) * DMODEL };
    for (int k = 0; k < DMODEL; k += 32) {
        __builtin_prefetch(Arow[0] + k + 128, 0, 3);
        __builtin_prefetch(Brow[0] + k + 128, 0, 3);
        __builtin_prefetch(Brow[2] + k + 128, 0, 3);
        bf16x16 a[2], b[4];
#pragma unroll
        for (int i = 0; i < 2; ++i) a[i] = load_a_frag(Arow[i] + k, DMODEL);
#pragma unroll
        for (int j = 0; j < 4; ++j) b[j] = load_b_frag(Brow[j] + k, DMODEL);
#pragma unroll
        for (int i = 0; i < 2; ++i)
#pragma unroll
            for (int j = 0; j < 4; ++j)
                c[i][j] = wmma_bf16(a[i], b[j], c[i][j]);
    }
#pragma unroll
    for (int i = 0; i < 2; ++i)
#pragma unroll
        for (int j = 0; j < 4; ++j)
            store_out_tile(c[i][j], m0 + i * 16, n0 + j * 16, bias, out);
}

// ---------------- host launch ----------------
extern "C" void kernel_launch(void* const* d_in, const int* in_sizes, int n_in,
                              void* d_out, int out_size, void* d_ws, size_t ws_size,
                              hipStream_t stream) {
    const float* x     = (const float*)d_in[0];   // [2,2048,2048]
    const float* wqkv  = (const float*)d_in[1];   // [6144,2048]
    const float* bqkv  = (const float*)d_in[2];   // [6144]
    const float* wproj = (const float*)d_in[3];   // [2048,2048]
    const float* bproj = (const float*)d_in[4];   // [2048]
    float* out = (float*)d_out;

    char* ws = (char*)d_ws;
    size_t off = 0;
    auto alloc = [&](size_t elems) {
        void* p = ws + off;
        off += (elems * sizeof(__bf16) + 255) & ~(size_t)255;
        return (__bf16*)p;
    };
    __bf16* xb     = alloc((size_t)NTOK * DMODEL);
    __bf16* wqkvb  = alloc((size_t)E3 * DMODEL);
    __bf16* wprojb = alloc((size_t)DMODEL * DMODEL);
    __bf16* Qb     = alloc((size_t)BB * NH * LL * DH);
    __bf16* Kb     = alloc((size_t)BB * NH * LL * DH);
    __bf16* Vtb    = alloc((size_t)BB * NH * DH * LL);
    __bf16* attno  = alloc((size_t)NTOK * DMODEL);

    const int nx = NTOK * DMODEL, nw = E3 * DMODEL, np = DMODEL * DMODEL;
    cvt_f32_bf16_kernel<<<(nx + 255) / 256, 256, 0, stream>>>(x, xb, nx);
    cvt_f32_bf16_kernel<<<(nw + 255) / 256, 256, 0, stream>>>(wqkv, wqkvb, nw);
    cvt_f32_bf16_kernel<<<(np + 255) / 256, 256, 0, stream>>>(wproj, wprojb, np);

    qkv_gemm_kernel<<<dim3(E3 / 256, NTOK / 32), 128, 0, stream>>>(
        xb, wqkvb, bqkv, Qb, Kb, Vtb);

    attn_kernel<<<dim3(LL / 16 / 4, NH, BB), 128, 0, stream>>>(
        Qb, Kb, Vtb, attno);

    proj_gemm_kernel<<<dim3(DMODEL / 256, NTOK / 32), 128, 0, stream>>>(
        attno, wprojb, bproj, out);
}
